// Light_PAM_64510408785997
// MI455X (gfx1250) — matrix operand order repaired and checked by reading, not compiled, not run on hardware
//
#include <hip/hip_runtime.h>

typedef __attribute__((ext_vector_type(16))) __bf16 v16bf;
typedef __attribute__((ext_vector_type(8)))  float  f32x8;
typedef __attribute__((ext_vector_type(4)))  unsigned int u32x4;

union AFrag {
    v16bf v;
    unsigned short h[16];
    u32x4 q[2];
};

static __device__ __forceinline__ unsigned short f2bf(float f) {
    unsigned int u = __builtin_bit_cast(unsigned int, f);
    u += 0x7fffu + ((u >> 16) & 1u);            // round-to-nearest-even
    return (unsigned short)(u >> 16);
}
static __device__ __forceinline__ float bf2f(unsigned short s) {
    unsigned int u = ((unsigned int)s) << 16;
    return __builtin_bit_cast(float, u);
}
static __device__ __forceinline__ f32x8 zero8() {
    f32x8 z = {0.f, 0.f, 0.f, 0.f, 0.f, 0.f, 0.f, 0.f};
    return z;
}
static __device__ __forceinline__ f32x8 wmma_bf16(const AFrag& a, const AFrag& b, f32x8 c) {
    return __builtin_amdgcn_wmma_f32_16x16x32_bf16(false, a.v, false, b.v, (short)0, c, false, false);
}

// A operand (M=16, K=32) from a row-major [row][ld] bf16 array (global or LDS).
// lane t: row = row0 + (t&15); K window {8h..8h+7} U {16+8h..16+8h+7}, h = t>>4.
static __device__ __forceinline__ AFrag load_a(const unsigned short* __restrict__ T,
                                               int row0, int k0, int ld, int lane) {
    AFrag a;
    const unsigned short* p = T + (size_t)(row0 + (lane & 15)) * ld + k0 + 8 * (lane >> 4);
    a.q[0] = *(const u32x4*)p;
    a.q[1] = *(const u32x4*)(p + 16);
    return a;
}
// B operand (K=32, N=16) from a *transposed* row-major [n][ld] bf16 array (n fastest dim = N).
// lane t: n = n0 + (t&15); K window {8h..8h+7} U {16+8h..16+8h+7}, h = t>>4.
static __device__ __forceinline__ AFrag load_b(const unsigned short* __restrict__ T,
                                               int n0, int k0, int ld, int lane) {
    AFrag b;
    const unsigned short* p = T + (size_t)(n0 + (lane & 15)) * ld + k0 + 8 * (lane >> 4);
    b.q[0] = *(const u32x4*)p;
    b.q[1] = *(const u32x4*)(p + 16);
    return b;
}

// ---------------------------------------------------------------------------
// geometry: N=8, C=256, C8=32, H=W=128, P=8, Q=16
// stage1: 512 problems, L=256 (16x16 of Q), batch b1=(n,ph,pw)
// stage2: 2048 problems, L=64 (8x8 of P), batch b2=(n,qh,qw)
// ---------------------------------------------------------------------------
template <int STAGE> struct Cfg {
    static constexpr int L    = (STAGE == 1) ? 256 : 64;
    static constexpr int BPW  = (STAGE == 1) ? 1 : 4;     // batches per workgroup
    static constexpr int LT   = L / 16;                   // l-tiles per batch (BPW*LT==16)
    static constexpr int XT_S = 256 + 8;                  // Xt row stride (bf16), mult of 8
    static constexpr int FD_S = L + 8;                    // FDc row stride, mult of 8
    static constexpr int XT_E = BPW * L * XT_S;
    static constexpr int FD_E = BPW * 256 * FD_S;
    static constexpr int FB_E = BPW * L * 32;
    static constexpr size_t SMEM = (size_t)(XT_E + FD_E + 2 * FB_E) * 2 + 320 * 4;
};

template <int STAGE>
static __device__ __forceinline__ size_t in_index(int blk, int bl, int c, int l) {
    if constexpr (STAGE == 1) {
        int n = blk >> 6, ph = (blk >> 3) & 7, pw = blk & 7;
        int qh = l >> 4, qw = l & 15;
        return ((size_t)((n * 256 + c) * 128 + qh * 8 + ph)) * 128 + (qw * 8 + pw);
    } else {
        int b2 = blk * 4 + bl;
        return ((size_t)(b2 * 256 + c)) * 64 + l;
    }
}
template <int STAGE>
static __device__ __forceinline__ size_t out_index(int blk, int bl, int c, int l) {
    if constexpr (STAGE == 1) {
        int n = blk >> 6, ph = (blk >> 3) & 7, pw = blk & 7;
        int qh = l >> 4, qw = l & 15;
        int b2 = (n << 8) + qh * 16 + qw;                 // stage-2 batch layout
        return ((size_t)(b2 * 256 + c)) * 64 + (ph * 8 + pw);
    } else {
        int b2 = blk * 4 + bl;
        int n = b2 >> 8, qh = (b2 >> 4) & 15, qw = b2 & 15;
        int ph = l >> 3, pw = l & 7;
        return ((size_t)((n * 256 + c) * 128 + qh * 8 + ph)) * 128 + (qw * 8 + pw);
    }
}

__global__ void convert_weights(const float* __restrict__ Wd, const float* __restrict__ Wb,
                                const float* __restrict__ Wc, unsigned short* __restrict__ o) {
    int i = blockIdx.x * 256 + threadIdx.x;               // 65536 threads total
    o[i] = f2bf(Wd[i]);
    if (i < 8192) {
        o[65536 + i] = f2bf(Wb[i]);
        o[73728 + i] = f2bf(Wc[i]);
    }
}

template <int STAGE>
__global__ __launch_bounds__(512, 1)
void pam_stage_k(const float* __restrict__ xin, const unsigned short* __restrict__ wpack,
                 const float* __restrict__ bb, const float* __restrict__ bc,
                 const float* __restrict__ bd, const float* __restrict__ alpha,
                 float* __restrict__ out) {
    using K = Cfg<STAGE>;
    constexpr int L = K::L, BPW = K::BPW, LT = K::LT;
    constexpr int XT_S = K::XT_S, FD_S = K::FD_S;

    extern __shared__ unsigned char smem[];
    unsigned short* Xt  = (unsigned short*)smem;          // [BPW][L][XT_S]  X transposed (B-ready)
    unsigned short* FDc = Xt + K::XT_E;                   // [BPW][256][FD_S] fd (B-ready for P@FD)
    unsigned short* FBt = FDc + K::FD_E;                  // [BPW][L][32]    fb^T (B-ready)
    unsigned short* FCt = FBt + K::FB_E;                  // [BPW][L][32]    fc^T (A-ready rows=m)
    float* bbS = (float*)(FCt + K::FB_E);
    float* bcS = bbS + 32;
    float* bdS = bcS + 32;
    unsigned short* Oc = Xt;                              // reuse: fe as [BPW][256][L] bf16

    const unsigned short* Wd_ = wpack;                    // [256][256]
    const unsigned short* Wb_ = wpack + 65536;            // [32][256]
    const unsigned short* Wc_ = wpack + 73728;            // [32][256]

    const int tid = threadIdx.x, lane = tid & 31, wave = tid >> 5;
    const int blk = blockIdx.x;

    // ---- phase 0: biases + gather X -> Xt (bf16, transposed) -------------
    if (tid < 32) bbS[tid] = bb[tid];
    else if (tid < 64) bcS[tid - 32] = bc[tid - 32];
    if (tid >= 64 && tid < 320) bdS[tid - 64] = bd[tid - 64];

    for (int e = tid; e < BPW * 256 * L; e += 512) {
        int bl = e / (256 * L), rem = e % (256 * L);
        int c = rem / L, l = rem % L;                     // l fastest: contiguous-ish global
        Xt[(size_t)(bl * L + l) * XT_S + c] = f2bf(xin[in_index<STAGE>(blk, bl, c, l)]);
    }
    __syncthreads();

    // ---- phase 1: FB/FC/FD GEMMs (D = W x X) -----------------------------
    // 20 tiles per (bl,lt) group: 2x FB o-tiles, 2x FC o-tiles, 16x FD c-tiles
    for (int t = wave; t < 16 * 20; t += 16) {
        int g = t / 20, u = t % 20;
        int bl = g / LT, lt = g % LT;
        const unsigned short* Wr;
        const float* bias;
        int row0;
        if (u < 2)       { Wr = Wb_; bias = bbS; row0 = u * 16; }
        else if (u < 4)  { Wr = Wc_; bias = bcS; row0 = (u - 2) * 16; }
        else             { Wr = Wd_; bias = bdS; row0 = (u - 4) * 16; }
        f32x8 acc = zero8();
        const unsigned short* xb = Xt + (size_t)bl * L * XT_S;
#pragma unroll
        for (int k0 = 0; k0 < 256; k0 += 32) {
            __builtin_prefetch(Wr + (size_t)(row0 + (lane & 15)) * 256 + k0 + 64, 0, 1);
            AFrag a = load_a(Wr, row0, k0, 256, lane);
            AFrag b = load_b(xb, lt * 16, k0, XT_S, lane);
            acc = wmma_bf16(a, b, acc);
        }
        const int nn = lane & 15, hh = lane >> 4;
#pragma unroll
        for (int r = 0; r < 8; ++r) {
            int row = row0 + r + 8 * hh;                  // o (or c for FD)
            int col = lt * 16 + nn;                       // l (or m for FD)
            unsigned short bv = f2bf(acc[r] + bias[row]);
            if (u < 2)      FBt[(size_t)(bl * L + col) * 32 + row] = bv;
            else if (u < 4) FCt[(size_t)(bl * L + col) * 32 + row] = bv;
            else            FDc[(size_t)(bl * 256 + row) * FD_S + col] = bv;
        }
    }
    __syncthreads();

    // ---- phase 2: attention (flash-style, S^T trick) ---------------------
    {
        const int bl = wave / LT, lt = wave % LT;
        const unsigned short* FBb = FBt + (size_t)bl * L * 32;
        const unsigned short* FCb = FCt + (size_t)bl * L * 32;
        const unsigned short* FDb = FDc + (size_t)bl * 256 * FD_S;

        AFrag Bl = load_b(FBb, lt * 16, 0, 32, lane);     // fb^T rows l (B operand, K=32)

        // pass 1: per-l row max + sum of exp (lane owns l = lt*16 + (lane&15))
        float rmax = -3.0e38f, rsum = 0.f;
        for (int m0 = 0; m0 < L; m0 += 16) {
            AFrag Ac = load_a(FCb, m0, 0, 32, lane);      // fc^T rows m (A operand)
            f32x8 s = wmma_bf16(Ac, Bl, zero8());         // S^T tile: M=m, N=l
            float tm = s[0];
#pragma unroll
            for (int r = 1; r < 8; ++r) tm = fmaxf(tm, s[r]);
            tm = fmaxf(tm, __shfl_xor(tm, 16, 32));
            float nm = fmaxf(rmax, tm);
            float ts = 0.f;
#pragma unroll
            for (int r = 0; r < 8; ++r) ts += __expf(s[r] - nm);
            ts += __shfl_xor(ts, 16, 32);
            rsum = rsum * __expf(rmax - nm) + ts;
            rmax = nm;
        }

        // pass 2: O[l][c] += P * FD ; S^T D-fragment == P A-fragment layout
        f32x8 O[16];
#pragma unroll
        for (int ct = 0; ct < 16; ++ct) O[ct] = zero8();
        for (int m0 = 0; m0 < L; m0 += 32) {
            f32x8 s1 = wmma_bf16(load_a(FCb, m0,      0, 32, lane), Bl, zero8());
            f32x8 s2 = wmma_bf16(load_a(FCb, m0 + 16, 0, 32, lane), Bl, zero8());
            AFrag P;
#pragma unroll
            for (int j = 0; j < 8; ++j) {
                P.h[j]     = f2bf(__expf(s1[j] - rmax));
                P.h[8 + j] = f2bf(__expf(s2[j] - rmax));
            }
#pragma unroll
            for (int ct = 0; ct < 16; ++ct) {
                AFrag Bd = load_b(FDb, ct * 16, m0, FD_S, lane);  // fd: K=m, N=c
                O[ct] = wmma_bf16(P, Bd, O[ct]);
            }
        }

        // normalize (broadcast 1/sum across the O fragment layout) and park in LDS
        float rinv = 1.0f / rsum;
        const int nn = lane & 15, hh = lane >> 4;
        float sc[8];
#pragma unroll
        for (int r = 0; r < 8; ++r) sc[r] = __shfl(rinv, r + 8 * hh, 32);
#pragma unroll
        for (int ct = 0; ct < 16; ++ct) {
#pragma unroll
            for (int r = 0; r < 8; ++r) {
                int l = lt * 16 + r + 8 * hh;
                int c = ct * 16 + nn;
                Oc[(size_t)(bl * 256 + c) * L + l] = f2bf(O[ct][r] * sc[r]);
            }
        }
    }
    __syncthreads();

    // ---- phase 3: out = alpha * fe + x (residual re-read in f32) ---------
    const float al = alpha[0];
    for (int e = tid; e < BPW * 256 * L; e += 512) {
        int bl = e / (256 * L), rem = e % (256 * L);
        int c = rem / L, l = rem % L;
        float fe = bf2f(Oc[(size_t)(bl * 256 + c) * L + l]);
        float res = xin[in_index<STAGE>(blk, bl, c, l)];
        out[out_index<STAGE>(blk, bl, c, l)] = al * fe + res;
    }
}

extern "C" void kernel_launch(void* const* d_in, const int* in_sizes, int n_in,
                              void* d_out, int out_size, void* d_ws, size_t ws_size,
                              hipStream_t stream) {
    const float* x   = (const float*)d_in[0];
    const float* Wb1 = (const float*)d_in[1];
    const float* bb1 = (const float*)d_in[2];
    const float* Wc1 = (const float*)d_in[3];
    const float* bc1 = (const float*)d_in[4];
    const float* Wd1 = (const float*)d_in[5];
    const float* bd1 = (const float*)d_in[6];
    const float* al1 = (const float*)d_in[7];
    const float* Wb2 = (const float*)d_in[8];
    const float* bb2 = (const float*)d_in[9];
    const float* Wc2 = (const float*)d_in[10];
    const float* bc2 = (const float*)d_in[11];
    const float* Wd2 = (const float*)d_in[12];
    const float* bd2 = (const float*)d_in[13];
    const float* al2 = (const float*)d_in[14];
    float* out = (float*)d_out;

    char* ws = (char*)d_ws;
    float* xmid = (float*)ws;                                  // 8*256*128*128 f32 = 128 MiB
    unsigned short* w1 = (unsigned short*)(ws + (size_t)134217728);
    unsigned short* w2 = w1 + 81920;

    convert_weights<<<256, 256, 0, stream>>>(Wd1, Wb1, Wc1, w1);
    convert_weights<<<256, 256, 0, stream>>>(Wd2, Wb2, Wc2, w2);

    size_t s1 = Cfg<1>::SMEM, s2 = Cfg<2>::SMEM;
    (void)hipFuncSetAttribute((const void*)&pam_stage_k<1>,
                              hipFuncAttributeMaxDynamicSharedMemorySize, (int)s1);
    (void)hipFuncSetAttribute((const void*)&pam_stage_k<2>,
                              hipFuncAttributeMaxDynamicSharedMemorySize, (int)s2);

    pam_stage_k<1><<<512, 512, s1, stream>>>(x, w1, bb1, bc1, bd1, al1, xmid);
    pam_stage_k<2><<<512, 512, s2, stream>>>(xmid, w2, bb2, bc2, bd2, al2, out);
}
